// UNet_39805756899533
// MI455X (gfx1250) — compile-verified, matrix operand order checked by reading, not run on hardware
//
#include <hip/hip_runtime.h>

typedef __bf16 bf16_t;
typedef __attribute__((ext_vector_type(16))) __bf16 v16bf;
typedef __attribute__((ext_vector_type(8)))  float  v8f;

#define BB   8
#define LL   4096
#define CC   256
#define HH   8
#define DD   32
#define CHK  128
#define NC   (LL / CHK)
#define MTOK (BB * LL)

static __device__ __forceinline__ v16bf ld16(const bf16_t* p) {
    return *(const v16bf*)p;
}
static __device__ __forceinline__ v8f wmma_bf16(v16bf a, v16bf b, v8f c) {
    return __builtin_amdgcn_wmma_f32_16x16x32_bf16(false, a, false, b, (short)0, c,
                                                   false, false);
}

// ---------------- elementwise prep ----------------
__global__ void k_prep(const float* __restrict__ x, const float* __restrict__ rel,
                       bf16_t* __restrict__ hbf) {
    int i = blockIdx.x * blockDim.x + threadIdx.x;
    if (i < MTOK * CC) hbf[i] = (bf16_t)(x[i] + rel[i % (LL * CC)]);
}

__global__ void k_cvt(const float* __restrict__ w, bf16_t* __restrict__ wb, int n) {
    int i = blockIdx.x * blockDim.x + threadIdx.x;
    if (i < n) wb[i] = (bf16_t)w[i];
}

// ---------------- QKV GEMM + fused RMS norm ----------------
// qkv[M=32768, N=768] = h[M,256] @ W^T ; N split as (3, H, 32)
__global__ __launch_bounds__(256) void k_qkv(const bf16_t* __restrict__ hbf,
                                             const bf16_t* __restrict__ wq,
                                             const float* __restrict__ qg,
                                             const float* __restrict__ kg,
                                             bf16_t* __restrict__ qb,
                                             bf16_t* __restrict__ kb,
                                             bf16_t* __restrict__ vb) {
    const int lane  = threadIdx.x & 31;
    const int wid   = blockIdx.x * (blockDim.x >> 5) + (threadIdx.x >> 5);
    const int mtile = wid / 24;
    const int npair = wid % 24;
    if (mtile >= MTOK / 16) return;

    const int r0 = lane & 15;
    const int kh = (lane >> 4) << 4;        // K sub-chunk per lane half
    const int n0 = npair * 32;

    const bf16_t* arow  = hbf + (size_t)(mtile * 16 + r0) * CC + kh;
    const bf16_t* b0row = wq + (size_t)(n0 + r0) * CC + kh;
    const bf16_t* b1row = wq + (size_t)(n0 + 16 + r0) * CC + kh;

    v8f acc0 = {}, acc1 = {};
#pragma unroll
    for (int kk = 0; kk < CC; kk += 32) {
        v16bf a  = ld16(arow + kk);
        v16bf b0 = ld16(b0row + kk);
        v16bf b1 = ld16(b1row + kk);
        acc0 = wmma_bf16(a, b0, acc0);
        acc1 = wmma_bf16(a, b1, acc1);
    }

    const int mat  = n0 >> 8;          // 0=q, 1=k, 2=v
    const int head = (n0 & 255) >> 5;

    if (mat == 0) {                    // q *= D^-0.5 before norm (eps-path fidelity)
        const float qs = 0.17677669529663687f;
#pragma unroll
        for (int r = 0; r < 8; ++r) { acc0[r] *= qs; acc1[r] *= qs; }
    }
    if (mat <= 1) {                    // fused RMS norm: row L2 over D=32, * sqrt(D)*gamma
        const float* g = (mat == 0) ? qg : kg;
        const float sg0 = 5.656854249492381f * g[head * DD + r0];
        const float sg1 = 5.656854249492381f * g[head * DD + 16 + r0];
#pragma unroll
        for (int r = 0; r < 8; ++r) {
            float p = acc0[r] * acc0[r] + acc1[r] * acc1[r];
            p += __shfl_xor(p, 1, 32);
            p += __shfl_xor(p, 2, 32);
            p += __shfl_xor(p, 4, 32);
            p += __shfl_xor(p, 8, 32);
            float inv = 1.0f / fmaxf(sqrtf(p), 1e-12f);
            acc0[r] *= inv * sg0;
            acc1[r] *= inv * sg1;
        }
    }

    bf16_t* dst = (mat == 0) ? qb : ((mat == 1) ? kb : vb);
#pragma unroll
    for (int r = 0; r < 8; ++r) {
        int tok = mtile * 16 + r + ((lane >> 4) << 3);
        int b   = tok >> 12;
        int lp  = tok & (LL - 1);
        bf16_t* rowp = dst + ((size_t)((b * HH + head) * LL) + lp) * DD;
        rowp[r0]      = (bf16_t)acc0[r];
        rowp[16 + r0] = (bf16_t)acc1[r];
    }
}

// ---------------- pass A: per-chunk KV contributions ----------------
// kvc[bh][n][e*32+d] = sum_t k[t,d]*exp(-s*(CHK-t)) * v[t,e]
__global__ __launch_bounds__(128) void k_kv(const bf16_t* __restrict__ kb,
                                            const bf16_t* __restrict__ vb,
                                            float* __restrict__ kvc) {
    __shared__ __attribute__((aligned(64))) bf16_t sKTd[DD * CHK];
    __shared__ __attribute__((aligned(64))) bf16_t sVT[DD * CHK];

    const int job = blockIdx.x;            // bh*NC + n, 2048 jobs
    const int bh  = job >> 5;
    const int n   = job & 31;
    const int h   = bh & 7;
    const int tid = threadIdx.x, lane = tid & 31, wv = tid >> 5;
    const float s = exp2f(-(float)(h + 1));

    const bf16_t* kcg = kb + ((size_t)bh * LL + n * CHK) * DD;
    const bf16_t* vcg = vb + ((size_t)bh * LL + n * CHK) * DD;
    for (int e = tid; e < CHK * DD; e += 128) {
        int t = e >> 5, d = e & 31;
        sKTd[d * CHK + t] = (bf16_t)((float)kcg[e] * __expf(-s * (float)(CHK - t)));
        sVT[d * CHK + t]  = vcg[e];
    }
    __syncthreads();

    const int dt = wv >> 1, et = wv & 1;   // 4 waves, one 16x16 tile each
    const int r0 = lane & 15, kh = (lane >> 4) << 4;
    v8f acc = {};
#pragma unroll
    for (int ks = 0; ks < 4; ++ks) {
        v16bf a  = ld16(&sKTd[(dt * 16 + r0) * CHK + ks * 32 + kh]);
        v16bf bb = ld16(&sVT[(et * 16 + r0) * CHK + ks * 32 + kh]);
        acc = wmma_bf16(a, bb, acc);
    }
    float* out = kvc + (size_t)job * (DD * DD);
#pragma unroll
    for (int r = 0; r < 8; ++r) {
        int d = dt * 16 + r + ((lane >> 4) << 3);
        int e = et * 16 + r0;
        out[e * DD + d] = acc[r];          // KV^T layout [e][d]
    }
}

// ---------------- pass B: in-place exclusive decay-scan over chunks ----------------
__global__ __launch_bounds__(256) void k_scan(float* __restrict__ kvc) {
    const int bh = blockIdx.x;             // 64 jobs
    const int h  = bh & 7;
    const float bdec = __expf(-exp2f(-(float)(h + 1)) * (float)CHK);
    float* base = kvc + (size_t)bh * NC * (DD * DD) + threadIdx.x * 4;
    float4 kv = make_float4(0.f, 0.f, 0.f, 0.f);
    for (int n = 0; n < NC; ++n) {
        float4* p = (float4*)(base + (size_t)n * (DD * DD));
        float4 c = *p;
        *p = kv;                           // KV seen by chunk n (pre-update)
        kv.x = kv.x * bdec + c.x;
        kv.y = kv.y * bdec + c.y;
        kv.z = kv.z * bdec + c.z;
        kv.w = kv.w * bdec + c.w;
    }
}

// ---------------- pass C: o = mask(q@k^T)@v + (q*qdecay)@KVpre ----------------
__global__ __launch_bounds__(256) void k_out(const bf16_t* __restrict__ qb,
                                             const bf16_t* __restrict__ kb,
                                             const bf16_t* __restrict__ vb,
                                             const float* __restrict__ kvc,
                                             bf16_t* __restrict__ ob) {
    __shared__ __attribute__((aligned(64))) bf16_t sQ[CHK * DD];
    __shared__ __attribute__((aligned(64))) bf16_t sQd[CHK * DD];
    __shared__ __attribute__((aligned(64))) bf16_t sK[CHK * DD];
    __shared__ __attribute__((aligned(64))) bf16_t sVT[DD * CHK];
    __shared__ __attribute__((aligned(64))) bf16_t sS[CHK * CHK];

    const int job = blockIdx.x;            // bh*NC + n, 2048 jobs
    const int bh  = job >> 5;
    const int n   = job & 31;
    const int b   = bh >> 3;
    const int h   = bh & 7;
    const int tid = threadIdx.x;
    const int lane = tid & 31;
    const int wv   = tid >> 5;             // 8 waves
    const int r0   = lane & 15;
    const int kh   = (lane >> 4) << 4;
    const float s  = exp2f(-(float)(h + 1));

    const bf16_t* qcg = qb + ((size_t)bh * LL + n * CHK) * DD;
    const bf16_t* kcg = kb + ((size_t)bh * LL + n * CHK) * DD;
    const bf16_t* vcg = vb + ((size_t)bh * LL + n * CHK) * DD;

    for (int e = tid; e < CHK * DD; e += 256) {
        int t = e >> 5, d = e & 31;
        bf16_t qv = qcg[e];
        sQ[e]  = qv;
        sQd[e] = (bf16_t)((float)qv * __expf(-s * (float)t));
        sK[e]  = kcg[e];
        sVT[d * CHK + t] = vcg[e];
    }
    __syncthreads();

    // scores = mask(q @ k^T); wave wv owns M-tile wv
    {
        const int it = wv;
        v16bf a = ld16(&sQ[(it * 16 + r0) * DD + kh]);
        for (int nt = 0; nt <= it; ++nt) {
            v16bf bfr = ld16(&sK[(nt * 16 + r0) * DD + kh]);
            v8f acc = {};
            acc = wmma_bf16(a, bfr, acc);
            const int jcol = nt * 16 + r0;
#pragma unroll
            for (int r = 0; r < 8; ++r) {
                int irow = it * 16 + r + ((lane >> 4) << 3);
                float df = (float)(irow - jcol);
                float m  = (df >= 0.f) ? __expf(-s * df) : 0.f;
                sS[irow * CHK + jcol] = (bf16_t)(acc[r] * m);
            }
        }
        for (int nt = it + 1; nt < 8; ++nt) {    // zero future tiles
            const int jcol = nt * 16 + r0;
#pragma unroll
            for (int r = 0; r < 8; ++r) {
                int irow = it * 16 + r + ((lane >> 4) << 3);
                sS[irow * CHK + jcol] = (bf16_t)0.f;
            }
        }
    }
    __syncthreads();

    // output tiles; wave wv owns M-tile wv, N-tiles 0..1
    {
        const int it = wv;
        const float* kvrow = kvc + (size_t)job * (DD * DD);
#pragma unroll
        for (int et = 0; et < 2; ++et) {
            v8f acc = {};
            {   // inter-chunk term from scanned KV (f32 -> bf16 fragment)
                v16bf a = ld16(&sQd[(it * 16 + r0) * DD + kh]);
                const float* kr = kvrow + (et * 16 + r0) * DD + kh;
                v16bf bb;
#pragma unroll
                for (int i = 0; i < 16; ++i) bb[i] = (bf16_t)kr[i];
                acc = wmma_bf16(a, bb, acc);
            }
            const int ksmax = it >> 1;           // causal: skip all-zero K-chunks
            for (int ks = 0; ks <= ksmax; ++ks) {
                v16bf a  = ld16(&sS[(it * 16 + r0) * CHK + ks * 32 + kh]);
                v16bf bb = ld16(&sVT[(et * 16 + r0) * CHK + ks * 32 + kh]);
                acc = wmma_bf16(a, bb, acc);
            }
#pragma unroll
            for (int r = 0; r < 8; ++r) {
                int irow = it * 16 + r + ((lane >> 4) << 3);
                int tok  = n * CHK + irow;
                int d    = et * 16 + r0;
                ob[((size_t)(b * LL + tok)) * CC + h * DD + d] = (bf16_t)acc[r];
            }
        }
    }
}

// ---------------- output projection ----------------
__global__ __launch_bounds__(256) void k_proj(const bf16_t* __restrict__ ab,
                                              const bf16_t* __restrict__ wp,
                                              const float* __restrict__ pb,
                                              float* __restrict__ out) {
    const int lane  = threadIdx.x & 31;
    const int wid   = blockIdx.x * (blockDim.x >> 5) + (threadIdx.x >> 5);
    const int mtile = wid / 8;
    const int npair = wid % 8;
    if (mtile >= MTOK / 16) return;

    const int r0 = lane & 15;
    const int kh = (lane >> 4) << 4;
    const int n0 = npair * 32;

    const bf16_t* arow  = ab + (size_t)(mtile * 16 + r0) * CC + kh;
    const bf16_t* b0row = wp + (size_t)(n0 + r0) * CC + kh;
    const bf16_t* b1row = wp + (size_t)(n0 + 16 + r0) * CC + kh;

    v8f acc0 = {}, acc1 = {};
#pragma unroll
    for (int kk = 0; kk < CC; kk += 32) {
        v16bf a  = ld16(arow + kk);
        v16bf b0 = ld16(b0row + kk);
        v16bf b1 = ld16(b1row + kk);
        acc0 = wmma_bf16(a, b0, acc0);
        acc1 = wmma_bf16(a, b1, acc1);
    }
    const float bias0 = pb[n0 + r0];
    const float bias1 = pb[n0 + 16 + r0];
#pragma unroll
    for (int r = 0; r < 8; ++r) {
        int row = mtile * 16 + r + ((lane >> 4) << 3);
        out[(size_t)row * CC + n0 + r0]      = acc0[r] + bias0;
        out[(size_t)row * CC + n0 + 16 + r0] = acc1[r] + bias1;
    }
}

extern "C" void kernel_launch(void* const* d_in, const int* in_sizes, int n_in,
                              void* d_out, int out_size, void* d_ws, size_t ws_size,
                              hipStream_t stream) {
    (void)in_sizes; (void)n_in; (void)out_size; (void)ws_size;
    const float* x    = (const float*)d_in[0];
    const float* rel  = (const float*)d_in[1];
    const float* qkvw = (const float*)d_in[2];
    const float* qg   = (const float*)d_in[3];
    const float* kg   = (const float*)d_in[4];
    const float* pw   = (const float*)d_in[5];
    const float* pbias= (const float*)d_in[6];
    float* out = (float*)d_out;

    char* ws = (char*)d_ws;
    const size_t SZ_ACT = (size_t)MTOK * CC * 2;   // 16 MiB bf16 activation plane
    bf16_t* hbf    = (bf16_t*)(ws);
    bf16_t* wqkvb  = (bf16_t*)(ws + SZ_ACT);
    bf16_t* wprojb = (bf16_t*)(ws + SZ_ACT + 393216);
    bf16_t* qb     = (bf16_t*)(ws + SZ_ACT + 393216 + 131072);
    bf16_t* kb     = (bf16_t*)((char*)qb + SZ_ACT);
    bf16_t* vb     = (bf16_t*)((char*)kb + SZ_ACT);
    bf16_t* ab     = (bf16_t*)((char*)vb + SZ_ACT);
    float*  kvc    = (float*)((char*)ab + SZ_ACT); // 64*32*1024 f32 = 8 MiB

    k_prep<<<(MTOK * CC) / 256, 256, 0, stream>>>(x, rel, hbf);
    k_cvt<<<(3 * CC * CC) / 256, 256, 0, stream>>>(qkvw, wqkvb, 3 * CC * CC);
    k_cvt<<<(CC * CC) / 256, 256, 0, stream>>>(pw, wprojb, CC * CC);
    k_qkv<<<(2048 * 24) / 8, 256, 0, stream>>>(hbf, wqkvb, qg, kg, qb, kb, vb);
    k_kv<<<BB * HH * NC, 128, 0, stream>>>(kb, vb, kvc);
    k_scan<<<BB * HH, 256, 0, stream>>>(kvc);
    k_out<<<BB * HH * NC, 256, 0, stream>>>(qb, kb, vb, kvc, ab);
    k_proj<<<(2048 * 8) / 8, 256, 0, stream>>>(ab, wprojb, pbias, out);
}
